// MambaBlock_56238301774002
// MI455X (gfx1250) — compile-verified, matrix operand order checked by reading, not run on hardware
//
#include <hip/hip_runtime.h>
#include <hip/hip_bf16.h>
#include <math.h>

typedef __attribute__((ext_vector_type(16))) __bf16 v16bf;
typedef __attribute__((ext_vector_type(8)))  __bf16 v8bf;
typedef __attribute__((ext_vector_type(4)))  __bf16 v4bf;
typedef __attribute__((ext_vector_type(8)))  float  v8f;

// ---------------------------------------------------------------------------
// fp32 -> bf16 bulk conversion (one-time, so the GEMM hot loop has zero cvt).
// ---------------------------------------------------------------------------
__global__ void __launch_bounds__(256)
f32_to_bf16_kernel(const float* __restrict__ in, __bf16* __restrict__ out, int n) {
    const int i = (blockIdx.x * blockDim.x + threadIdx.x) * 8;
    if (i + 7 >= n) {
        for (int j = i; j < n; ++j) out[j] = (__bf16)in[j];
        return;
    }
    float4 a = *(const float4*)(in + i);
    float4 b = *(const float4*)(in + i + 4);
    v4bf lo, hi;
    lo[0] = (__bf16)a.x; lo[1] = (__bf16)a.y; lo[2] = (__bf16)a.z; lo[3] = (__bf16)a.w;
    hi[0] = (__bf16)b.x; hi[1] = (__bf16)b.y; hi[2] = (__bf16)b.z; hi[3] = (__bf16)b.w;
    *(v4bf*)(out + i)     = lo;
    *(v4bf*)(out + i + 4) = hi;
}

// ---------------------------------------------------------------------------
// bf16 fragment load: CDNA5 wave32 16-bit A/B layout (ISA 7.12.2).
//   lanes 0-15  : row = base+lane,    elems 0..7 = K{0..7},  8..15 = K{16..23}
//   lanes 16-31 : row = base+lane-16, elems 0..7 = K{8..15}, 8..15 = K{24..31}
// p already points at (row, k0 + khalf*8): two contiguous 16-byte loads.
// ---------------------------------------------------------------------------
__device__ __forceinline__ v16bf load_frag(const __bf16* __restrict__ p) {
    v8bf lo = *(const v8bf*)(p);
    v8bf hi = *(const v8bf*)(p + 16);
    return __builtin_shufflevector(lo, hi, 0, 1, 2, 3, 4, 5, 6, 7,
                                           8, 9, 10, 11, 12, 13, 14, 15);
}

// ---------------------------------------------------------------------------
// out[m, n] = act( sum_k A[m,k] * W[n,k] + bias[n] ),  A/W bf16, out fp32.
// Each wave computes a 32x64 block: 2 A-fragments x 4 B-fragments
// -> 8 v_wmma_f32_16x16x32_bf16 per K-step (0.75 fragment loads per WMMA).
// Block = 8 waves.  grid.x = (M/32)*(N/64)/8.  act: 0 = none, 1 = relu.
// ---------------------------------------------------------------------------
__global__ void __launch_bounds__(256)
gemm_bf16_wmma_kernel(const __bf16* __restrict__ A, const __bf16* __restrict__ W,
                      const float* __restrict__ bias, float* __restrict__ out,
                      int M, int N, int K, int act) {
    const int lane    = threadIdx.x & 31;
    const int wave    = threadIdx.x >> 5;
    const int nGroups = N >> 6;                      // 64-wide strips
    const int tile    = blockIdx.x * 8 + wave;
    const int mTile   = tile / nGroups;              // 32-row blocks
    const int nGroup  = tile - mTile * nGroups;
    if (mTile >= (M >> 5)) return;

    const int m0    = mTile << 5;
    const int n0    = nGroup << 6;
    const int l15   = lane & 15;
    const int khalf = lane >> 4;

    const __bf16* pa = A + (size_t)(m0 + l15) * K + khalf * 8;
    const __bf16* pb = W + (size_t)(n0 + l15) * K + khalf * 8;
    const size_t  rStride = (size_t)16 * K;          // 16 rows

    v8f acc00 = {}, acc01 = {}, acc02 = {}, acc03 = {};
    v8f acc10 = {}, acc11 = {}, acc12 = {}, acc13 = {};

    for (int k0 = 0; k0 < K; k0 += 32) {
        if (k0 + 32 < K) {
            __builtin_prefetch(pa + k0 + 32, 0, 3);            // gfx1250 global_prefetch_b8
            __builtin_prefetch(pa + rStride + k0 + 32, 0, 3);
        }
        v16bf a0 = load_frag(pa + k0);
        v16bf a1 = load_frag(pa + rStride + k0);
        v16bf b0 = load_frag(pb + k0);
        v16bf b1 = load_frag(pb + rStride + k0);
        v16bf b2 = load_frag(pb + 2 * rStride + k0);
        v16bf b3 = load_frag(pb + 3 * rStride + k0);
        acc00 = __builtin_amdgcn_wmma_f32_16x16x32_bf16(false, a0, false, b0, (short)0, acc00, false, false);
        acc01 = __builtin_amdgcn_wmma_f32_16x16x32_bf16(false, a0, false, b1, (short)0, acc01, false, false);
        acc02 = __builtin_amdgcn_wmma_f32_16x16x32_bf16(false, a0, false, b2, (short)0, acc02, false, false);
        acc03 = __builtin_amdgcn_wmma_f32_16x16x32_bf16(false, a0, false, b3, (short)0, acc03, false, false);
        acc10 = __builtin_amdgcn_wmma_f32_16x16x32_bf16(false, a1, false, b0, (short)0, acc10, false, false);
        acc11 = __builtin_amdgcn_wmma_f32_16x16x32_bf16(false, a1, false, b1, (short)0, acc11, false, false);
        acc12 = __builtin_amdgcn_wmma_f32_16x16x32_bf16(false, a1, false, b2, (short)0, acc12, false, false);
        acc13 = __builtin_amdgcn_wmma_f32_16x16x32_bf16(false, a1, false, b3, (short)0, acc13, false, false);
    }

    // C/D layout: lanes 0-15 -> N=lane, M=mBase+r; lanes 16-31 -> M=mBase+8+r.
    v8f accs[2][4] = {{acc00, acc01, acc02, acc03}, {acc10, acc11, acc12, acc13}};
    #pragma unroll
    for (int i = 0; i < 2; ++i) {
        const int rowBase = m0 + i * 16 + khalf * 8;
        #pragma unroll
        for (int j = 0; j < 4; ++j) {
            const int   col = n0 + j * 16 + l15;
            const float bv  = bias ? bias[col] : 0.0f;
            #pragma unroll
            for (int r = 0; r < 8; ++r) {
                float v = accs[i][j][r] + bv;
                if (act == 1) v = fmaxf(v, 0.0f);
                out[(size_t)(rowBase + r) * N + col] = v;
            }
        }
    }
}

// ---------------------------------------------------------------------------
// Fused selective-scan + gating. One thread per (b, n) channel, serial over T.
//   h = exp(d*A_sel)*h + d*B_sel*z ;  y = C_sel*h ; if type==0: y *= sigmoid(g)
// Reads z/gate fp32, writes y as bf16 (feeds the bf16 output GEMM directly).
// ---------------------------------------------------------------------------
__global__ void __launch_bounds__(256)
mamba_scan_kernel(const float* __restrict__ z, const float* __restrict__ gate,
                  __bf16* __restrict__ ybf, const int* __restrict__ type_ids,
                  const float* __restrict__ A1, const float* __restrict__ A2,
                  const float* __restrict__ B1, const float* __restrict__ B2,
                  const float* __restrict__ C1, const float* __restrict__ C2,
                  const float* __restrict__ delta,
                  int Bn, int T, int N) {
    const int idx = blockIdx.x * blockDim.x + threadIdx.x;
    if (idx >= Bn * N) return;
    const int b = idx / N;
    const int n = idx - b * N;

    float h = 0.0f;
    for (int t = 0; t < T; ++t) {
        const int    ty  = type_ids[b * T + t];
        const size_t tn  = (size_t)t * N + n;
        const size_t btn = ((size_t)b * T + t) * N + n;
        const bool coarse = (ty == 1);
        const float a  = coarse ? A1[tn] : A2[tn];
        const float bb = coarse ? B1[tn] : B2[tn];
        const float c  = coarse ? C1[tn] : C2[tn];
        const float d  = delta[t];
        const float zt = z[btn];

        h = __expf(d * a) * h + (d * bb) * zt;
        float y = c * h;
        if (ty == 0) {
            const float g = gate[btn];
            y *= 1.0f / (1.0f + __expf(-g));
        }
        ybf[btn] = (__bf16)y;
    }
}

// ---------------------------------------------------------------------------
// Launcher. Inputs (setup_inputs order):
//  0:x 1:type_ids 2:Wg 3:bg 4:Ws 5:bs 6:A1 7:A2 8:B1 9:B2 10:C1 11:C2
//  12:delta 13:Wo 14:bo
// Workspace: [z fp32 | gate fp32 | xbf (reused as ybf) | Wg_bf | Ws_bf | Wo_bf]
// ---------------------------------------------------------------------------
extern "C" void kernel_launch(void* const* d_in, const int* in_sizes, int n_in,
                              void* d_out, int out_size, void* d_ws, size_t ws_size,
                              hipStream_t stream) {
    const float* x        = (const float*)d_in[0];
    const int*   type_ids = (const int*)  d_in[1];
    const float* Wg       = (const float*)d_in[2];
    const float* bg       = (const float*)d_in[3];
    const float* Ws       = (const float*)d_in[4];
    const float* bs       = (const float*)d_in[5];
    const float* A1       = (const float*)d_in[6];
    const float* A2       = (const float*)d_in[7];
    const float* B1       = (const float*)d_in[8];
    const float* B2       = (const float*)d_in[9];
    const float* C1       = (const float*)d_in[10];
    const float* C2       = (const float*)d_in[11];
    const float* delta    = (const float*)d_in[12];
    const float* Wo       = (const float*)d_in[13];
    const float* bo       = (const float*)d_in[14];
    float*       out      = (float*)d_out;

    const int T  = in_sizes[12];                 // 2048
    const int BT = in_sizes[1];                  // B*T
    const int Bn = BT / T;                       // 8
    const int D  = in_sizes[0] / BT;             // 1024
    const int N  = in_sizes[2] / D;              // 1024
    const int M  = BT;                           // 16384 GEMM rows

    const size_t MN = (size_t)M * N;
    const size_t MD = (size_t)M * D;
    const size_t ND = (size_t)N * D;

    float*  zbuf = (float*)d_ws;                 // MN fp32
    float*  gbuf = zbuf + MN;                    // MN fp32
    __bf16* xbf  = (__bf16*)(gbuf + MN);         // MD bf16 (reused as ybf)
    __bf16* ybf  = xbf;                          // dead x after GEMMs 1&2
    __bf16* wgbf = xbf + MD;                     // ND bf16
    __bf16* wsbf = wgbf + ND;                    // ND bf16
    __bf16* wobf = wsbf + ND;                    // ND bf16

    const dim3 blk(256);

    // One-time bf16 casts (8 elems/thread).
    f32_to_bf16_kernel<<<dim3((MD / 8 + 255) / 256), blk, 0, stream>>>(x, xbf, (int)MD);
    f32_to_bf16_kernel<<<dim3((ND / 8 + 255) / 256), blk, 0, stream>>>(Wg, wgbf, (int)ND);
    f32_to_bf16_kernel<<<dim3((ND / 8 + 255) / 256), blk, 0, stream>>>(Ws, wsbf, (int)ND);
    f32_to_bf16_kernel<<<dim3((ND / 8 + 255) / 256), blk, 0, stream>>>(Wo, wobf, (int)ND);

    const int  strips = (M >> 5) * (N >> 6);     // 32x64 wave blocks
    const dim3 grd((strips + 7) / 8);

    // z = relu(x @ Ws^T + bs) ; gate = relu(x @ Wg^T + bg)
    gemm_bf16_wmma_kernel<<<grd, blk, 0, stream>>>(xbf, wsbf, bs, zbuf, M, N, D, 1);
    gemm_bf16_wmma_kernel<<<grd, blk, 0, stream>>>(xbf, wgbf, bg, gbuf, M, N, D, 1);

    // fused selective scan + sigmoid gating -> y (bf16)
    const int scanThreads = Bn * N;              // 8192
    mamba_scan_kernel<<<dim3((scanThreads + 255) / 256), blk, 0, stream>>>(
        zbuf, gbuf, ybf, type_ids, A1, A2, B1, B2, C1, C2, delta, Bn, T, N);

    // out = y @ Wo^T + bo
    const int  stripsO = (M >> 5) * (D >> 6);
    const dim3 grdO((stripsO + 7) / 8);
    gemm_bf16_wmma_kernel<<<grdO, blk, 0, stream>>>(ybf, wobf, bo, out, M, D, N, 0);
}
